// MLALayer_19662360281111
// MI455X (gfx1250) — compile-verified, matrix operand order checked by reading, not run on hardware
//
#include <hip/hip_runtime.h>
#include <hip/hip_bf16.h>

// ---------------------------------------------------------------------------
// MLA layer for MI455X (gfx1250, wave32, WMMA).
// Pipeline: convert->f16, 3 projection GEMMs, flash-attention, output GEMM.
// All matrix math via v_wmma_f32_16x16x32_f16; all hot-path memory ops forced
// to addrspace(1) so they lower to global_load/global_store (LOADcnt only),
// not flat_* (which would also tie up DScnt / the LDS path).
// ---------------------------------------------------------------------------

typedef _Float16 v16h __attribute__((ext_vector_type(16)));
typedef _Float16 v8h  __attribute__((ext_vector_type(8)));
typedef float    v8f  __attribute__((ext_vector_type(8)));

typedef const v8h      __attribute__((address_space(1)))* gv8h_cp;
typedef const _Float16 __attribute__((address_space(1)))* gh_cp;
typedef _Float16       __attribute__((address_space(1)))* gh_p;
typedef float          __attribute__((address_space(1)))* gf_p;

#define WMMA_F16(a, b, c) \
    __builtin_amdgcn_wmma_f32_16x16x32_f16(false, (a), false, (b), (short)0, (c), false, false)

// Load a 16x32 (A) or 32x16 (B) f16 WMMA fragment for this lane.
// Layout (ISA 7.12.2): lane l holds row/col (l&15), K = (l>>4)*8 + {0..7, 16..23}.
// With K-contiguous storage this is two global_load_b128.
__device__ __forceinline__ v16h ld_frag(const _Float16* p) {
    gv8h_cp g = (gv8h_cp)(unsigned long long)p;
    v8h lo = g[0];   // K = base + 0..7
    v8h hi = g[2];   // K = base + 16..23
    return __builtin_shufflevector(lo, hi, 0, 1, 2, 3, 4, 5, 6, 7,
                                           8, 9, 10, 11, 12, 13, 14, 15);
}

__device__ __forceinline__ _Float16 ldh(const _Float16* p) {
    return *(gh_cp)(unsigned long long)p;
}
__device__ __forceinline__ void sth(_Float16* p, _Float16 v) {
    *(gh_p)(unsigned long long)p = v;
}
__device__ __forceinline__ void stf(float* p, float v) {
    *(gf_p)(unsigned long long)p = v;
}

__device__ __forceinline__ float lane_read(int src_lane, float v) {
    return __builtin_bit_cast(
        float, __builtin_amdgcn_ds_bpermute(src_lane << 2, __builtin_bit_cast(int, v)));
}

// ---------------------------------------------------------------------------
// Elementwise f32 -> f16 conversion.
// ---------------------------------------------------------------------------
__global__ void k_f32_to_f16(const float* __restrict__ in, _Float16* __restrict__ out, int n) {
    int i = blockIdx.x * blockDim.x + threadIdx.x;
    int stride = gridDim.x * blockDim.x;
    for (; i < n; i += stride) out[i] = (_Float16)in[i];
}

// Transpose-convert weight [K][N] f32 -> [N][K] f16 (K = contraction dim).
__global__ void k_transpose_f16(const float* __restrict__ in, _Float16* __restrict__ out,
                                int K, int N) {
    int i = blockIdx.x * blockDim.x + threadIdx.x;
    int total = K * N;
    int stride = gridDim.x * blockDim.x;
    for (; i < total; i += stride) {
        int n = i / K, k = i - n * K;
        out[i] = (_Float16)in[(size_t)k * N + n];
    }
}

// ---------------------------------------------------------------------------
// GEMM: C[M,N] = A[M,K] * B[K,N], A f16 row-major, Bt f16 [N][K] (transposed).
// Block = 128 threads = 4 waves in 2x2; each wave computes a 64x64 tile
// (4x4 grid of 16x16 WMMA tiles). Requires M%128==0, N%128==0, K%32==0.
// launch_bounds(128,1): full VGPR budget, no accumulator spilling.
// ---------------------------------------------------------------------------
__global__ __launch_bounds__(128, 1) void k_gemm_wmma(
        const _Float16* __restrict__ A, const _Float16* __restrict__ Bt,
        void* __restrict__ C, int M, int N, int K, int store_half) {
    const int wave = threadIdx.x >> 5;
    const int lane = threadIdx.x & 31;
    const int lr   = lane & 15;
    const int hi8  = (lane >> 4) << 3;

    const int row0 = blockIdx.y * 128 + (wave >> 1) * 64;
    const int col0 = blockIdx.x * 128 + (wave & 1) * 64;

    // Hoisted per-tile row pointers; advanced by 32 halves per K step.
    const _Float16* pa[4];
    const _Float16* pb[4];
#pragma unroll
    for (int t = 0; t < 4; ++t) {
        pa[t] = A + (size_t)(row0 + t * 16 + lr) * K + hi8;
        pb[t] = Bt + (size_t)(col0 + t * 16 + lr) * K + hi8;
    }

    v8f acc[4][4] = {};

    for (int k0 = 0; k0 < K; k0 += 32) {
        v16h af[4];
#pragma unroll
        for (int mt = 0; mt < 4; ++mt) {
            af[mt] = ld_frag(pa[mt]);
            pa[mt] += 32;
        }
#pragma unroll
        for (int nt = 0; nt < 4; ++nt) {
            v16h bf = ld_frag(pb[nt]);
            pb[nt] += 32;
#pragma unroll
            for (int mt = 0; mt < 4; ++mt)
                acc[mt][nt] = WMMA_F16(af[mt], bf, acc[mt][nt]);
        }
    }

    // D layout: lane holds col = (tile)+lr, rows = (tile) + hi8 + {0..7}.
    if (store_half) {
        _Float16* Ch = (_Float16*)C;
#pragma unroll
        for (int mt = 0; mt < 4; ++mt)
#pragma unroll
            for (int nt = 0; nt < 4; ++nt) {
                int col = col0 + nt * 16 + lr;
#pragma unroll
                for (int i = 0; i < 8; ++i)
                    sth(Ch + (size_t)(row0 + mt * 16 + hi8 + i) * N + col,
                        (_Float16)acc[mt][nt][i]);
            }
    } else {
        float* Cf = (float*)C;
#pragma unroll
        for (int mt = 0; mt < 4; ++mt)
#pragma unroll
            for (int nt = 0; nt < 4; ++nt) {
                int col = col0 + nt * 16 + lr;
#pragma unroll
                for (int i = 0; i < 8; ++i)
                    stf(Cf + (size_t)(row0 + mt * 16 + hi8 + i) * N + col, acc[mt][nt][i]);
            }
    }
}

// ---------------------------------------------------------------------------
// Causal flash attention. One wave owns one 16-row q-tile of one (b,h).
// Trick: compute S^T = K_blk x Q^T so the WMMA D-layout of S^T (lane = q-row,
// regs = k-positions) is exactly the A-fragment layout for the P x V WMMA.
// Q layout: [B*T, 1024] f16 (col h*64+d). KV layout: [B*T, 2048] f16
// (K at col h*64+d, V at 1024+h*64+d). Y out: [B*T, 1024] f16.
// ---------------------------------------------------------------------------
__global__ __launch_bounds__(256, 1) void k_attention(
        const _Float16* __restrict__ Q, const _Float16* __restrict__ KV,
        _Float16* __restrict__ Y) {
    const int wid = blockIdx.x * 8 + (threadIdx.x >> 5);
    const int qt = wid & 15;           // q tile (16 rows)
    const int h  = (wid >> 4) & 15;    // head
    const int b  = wid >> 8;           // batch

    const int lane = threadIdx.x & 31;
    const int lr   = lane & 15;
    const int hi8  = (lane >> 4) << 3;

    const int qrow = qt * 16 + lr;     // this lane's q row (0..255)
    const float NEG_INF = -__builtin_inff();

    // Q^T fragments (B operand): lane col = q-row lr; K dim = head dim.
    const _Float16* qp = Q + (size_t)(b * 256 + qrow) * 1024 + h * 64;
    const v16h bq0 = ld_frag(qp + hi8);        // d = 0..31
    const v16h bq1 = ld_frag(qp + 32 + hi8);   // d = 32..63

    const _Float16* kbase = KV + (size_t)(b * 256) * 2048 + h * 64;
    // Lane-resident base pointers; bumped by 32 rows per k-block so all row
    // offsets fold into the 24-bit instruction immediate.
    const _Float16* kptr = kbase + (size_t)lr * 2048 + hi8;          // A frag rows
    const _Float16* vptr = kbase + 1024 + (size_t)hi8 * 2048 + lr;   // B frag cols

    float m_run = NEG_INF, l_run = 0.0f;
    v8f acc[4] = {};

    const int nkb = ((qt * 16 + 15) >> 5) + 1;   // 32-wide k blocks to cover causal span
    for (int kb = 0; kb < nkb; ++kb) {
        const int kpos0 = kb * 32;

        // Two 16x16 S^T tiles covering k positions kpos0..kpos0+31.
        v8f st[2];
#pragma unroll
        for (int t2 = 0; t2 < 2; ++t2) {
            v16h ak0 = ld_frag(kptr + t2 * 16 * 2048);        // d 0..31
            v16h ak1 = ld_frag(kptr + t2 * 16 * 2048 + 32);   // d 32..63
            v8f s = {};
            s = WMMA_F16(ak0, bq0, s);
            s = WMMA_F16(ak1, bq1, s);
            st[t2] = s;
        }

        // Scale, causal mask, lane-local max (lane holds 16 k-positions of row qrow).
        float mt = NEG_INF;
#pragma unroll
        for (int t2 = 0; t2 < 2; ++t2)
#pragma unroll
            for (int i = 0; i < 8; ++i) {
                float v = st[t2][i] * 0.125f;                 // HEAD_SIZE^-0.5
                int kpos = kpos0 + t2 * 16 + hi8 + i;
                v = (kpos <= qrow) ? v : NEG_INF;
                st[t2][i] = v;
                mt = fmaxf(mt, v);
            }
        mt = fmaxf(mt, lane_read(lane ^ 16, mt));             // other 16 k-positions
        const float m_new = fmaxf(m_run, mt);
        const float corr  = __expf(m_run - m_new);

        // p = exp(s - m_new); pack directly into the A fragment for P x V.
        v16h ap;
        float rs = 0.0f;
#pragma unroll
        for (int i = 0; i < 8; ++i) {
            float p = __expf(st[0][i] - m_new);
            rs += p;
            ap[i] = (_Float16)p;
        }
#pragma unroll
        for (int i = 0; i < 8; ++i) {
            float p = __expf(st[1][i] - m_new);
            rs += p;
            ap[8 + i] = (_Float16)p;
        }
        rs += lane_read(lane ^ 16, rs);
        l_run = l_run * corr + rs;
        m_run = m_new;

        // Broadcast per-row rescale factors to the Y-accumulator layout
        // (acc rows = q-rows hi8..hi8+7; corr for q-row r lives in lane r).
        float cf[8];
#pragma unroll
        for (int i = 0; i < 8; ++i) cf[i] = lane_read(hi8 + i, corr);

        // Y += P x V over 4 head-dim tiles; all V loads share one base pointer
        // (row offsets fold into the instruction immediate).
#pragma unroll
        for (int dt = 0; dt < 4; ++dt) {
            v16h bv;
#pragma unroll
            for (int j = 0; j < 8; ++j)
                bv[j] = ldh(vptr + j * 2048 + dt * 16);
#pragma unroll
            for (int j = 0; j < 8; ++j)
                bv[8 + j] = ldh(vptr + (16 + j) * 2048 + dt * 16);
            v8f c;
#pragma unroll
            for (int i = 0; i < 8; ++i) c[i] = acc[dt][i] * cf[i];
            acc[dt] = WMMA_F16(ap, bv, c);
        }

        kptr += 32 * 2048;
        vptr += 32 * 2048;
    }

    // Normalize and store Y (f16, [B*T, 1024], col h*64+d).
    const float li = 1.0f / l_run;
    float inv[8];
#pragma unroll
    for (int i = 0; i < 8; ++i) inv[i] = lane_read(hi8 + i, li);
    _Float16* yp = Y + (size_t)(b * 256 + qt * 16 + hi8) * 1024 + h * 64 + lr;
#pragma unroll
    for (int dt = 0; dt < 4; ++dt)
#pragma unroll
        for (int i = 0; i < 8; ++i) {
            float y = acc[dt][i] * inv[i];
            sth(yp + (size_t)i * 1024 + dt * 16, (_Float16)y);
        }
}

// ---------------------------------------------------------------------------
// Host-side orchestration.
// ---------------------------------------------------------------------------
extern "C" void kernel_launch(void* const* d_in, const int* in_sizes, int n_in,
                              void* d_out, int out_size, void* d_ws, size_t ws_size,
                              hipStream_t stream) {
    (void)in_sizes; (void)n_in; (void)out_size; (void)ws_size;

    const float* x        = (const float*)d_in[0];   // [64,256,1024]
    const float* w_kvdown = (const float*)d_in[1];   // [1024,128]
    const float* w_kvup   = (const float*)d_in[2];   // [128,2048]
    const float* w_q      = (const float*)d_in[3];   // [1024,1024]
    const float* w_out    = (const float*)d_in[4];   // [1024,1024]
    float* out            = (float*)d_out;           // [64,256,1024]

    const int BT = 64 * 256;   // 16384
    const int C  = 1024;
    const int HD = 1024;
    const int R  = 128;        // KV_RANK
    const int KVW = 2048;

    // Carve f16 workspace (total ~177 MB).
    size_t off = 0;
    auto carve = [&](size_t halves) {
        _Float16* p = (_Float16*)((char*)d_ws + off);
        off = (off + halves * sizeof(_Float16) + 255) & ~(size_t)255;
        return p;
    };
    _Float16* xh     = carve((size_t)BT * C);
    _Float16* wqT    = carve((size_t)HD * C);      // [N=1024][K=1024]
    _Float16* wkvdT  = carve((size_t)R * C);       // [N=128][K=1024]
    _Float16* wkvupT = carve((size_t)KVW * R);     // [N=2048][K=128]
    _Float16* woutT  = carve((size_t)C * HD);      // [N=1024][K=1024]
    _Float16* qh     = carve((size_t)BT * HD);
    _Float16* lh     = carve((size_t)BT * R);
    _Float16* kvh    = carve((size_t)BT * KVW);
    _Float16* yh     = carve((size_t)BT * HD);

    // 1) Convert inputs to f16 (weights transposed to [N][K]).
    k_f32_to_f16<<<4096, 256, 0, stream>>>(x, xh, BT * C);
    k_transpose_f16<<<2048, 256, 0, stream>>>(w_q,      wqT,    C, HD);
    k_transpose_f16<<<1024, 256, 0, stream>>>(w_kvdown, wkvdT,  C, R);
    k_transpose_f16<<<1024, 256, 0, stream>>>(w_kvup,   wkvupT, R, KVW);
    k_transpose_f16<<<2048, 256, 0, stream>>>(w_out,    woutT,  C, HD);

    // 2) Projections.
    k_gemm_wmma<<<dim3(HD / 128, BT / 128), 128, 0, stream>>>(xh, wqT, qh, BT, HD, C, 1);
    k_gemm_wmma<<<dim3(R / 128, BT / 128), 128, 0, stream>>>(xh, wkvdT, lh, BT, R, C, 1);
    k_gemm_wmma<<<dim3(KVW / 128, BT / 128), 128, 0, stream>>>(lh, wkvupT, kvh, BT, KVW, R, 1);

    // 3) Attention: 64*16*16 = 16384 waves, 8 waves/block.
    k_attention<<<2048, 256, 0, stream>>>(qh, kvh, yh);

    // 4) Output projection (f32 result).
    k_gemm_wmma<<<dim3(C / 128, BT / 128), 128, 0, stream>>>(yh, woutT, out, BT, C, HD, 0);
}
